// KVCachedMultiHeadAttention_64905545777621
// MI455X (gfx1250) — matrix operand order repaired and checked by reading, not compiled, not run on hardware
//
#include <hip/hip_runtime.h>
#include <stdint.h>
#include <math.h>

typedef __attribute__((ext_vector_type(16))) _Float16 v16h;
typedef __attribute__((ext_vector_type(8)))  _Float16 v8h;
typedef __attribute__((ext_vector_type(8)))  float    v8f;

#define D_MODEL   2048
#define NUM_HEADS 16
#define HEAD_DIM  128
#define BATCH     4
#define SEQ_Q     1024
#define SEQ_C     1024
#define SKV       (SEQ_C + SEQ_Q)

__device__ __forceinline__ v16h cat8(v8h a, v8h b) {
  return __builtin_shufflevector(a, b, 0,1,2,3,4,5,6,7,8,9,10,11,12,13,14,15);
}

// ---------------------------------------------------------------------
// CDNA5 async global->LDS copies (ASYNCcnt tracked) via inline asm.
// ---------------------------------------------------------------------
__device__ __forceinline__ uint32_t lds_off(const void* p) {
  return (uint32_t)(uintptr_t)p;
}
__device__ __forceinline__ void async_b128(uint32_t ldst, const void* g) {
  asm volatile("global_load_async_to_lds_b128 %0, %1, off"
               :: "v"(ldst), "v"(g) : "memory");
}
__device__ __forceinline__ void async_b64(uint32_t ldst, const void* g) {
  asm volatile("global_load_async_to_lds_b64 %0, %1, off"
               :: "v"(ldst), "v"(g) : "memory");
}
__device__ __forceinline__ void wait_async0() {
  asm volatile("s_wait_asynccnt 0x0" ::: "memory");
}

// =====================================================================
// One-time fp32 -> f16 conversion (weights / K cache).
// =====================================================================
__global__ __launch_bounds__(256) void cvt_f32_to_f16(
    const float* __restrict__ x, _Float16* __restrict__ y, int n)
{
  const int i = (blockIdx.x * 256 + threadIdx.x) * 8;
  if (i >= n) return;
  const float4 a = *(const float4*)(x + i);
  const float4 b = *(const float4*)(x + i + 4);
  v8h o;
  o[0] = (_Float16)a.x; o[1] = (_Float16)a.y;
  o[2] = (_Float16)a.z; o[3] = (_Float16)a.w;
  o[4] = (_Float16)b.x; o[5] = (_Float16)b.y;
  o[6] = (_Float16)b.z; o[7] = (_Float16)b.w;
  *(v8h*)(y + i) = o;
}

// =====================================================================
// One-time cached_v [B,SC,D] fp32 -> Vt [B][D][SKV] f16 (kv-major).
// Reads coalesced along c; each lane packs 8 consecutive kv -> one b128.
// grid = (D/256, B*SC/8), block = 256.
// =====================================================================
__global__ __launch_bounds__(256) void cvt_transpose_v(
    const float* __restrict__ src, _Float16* __restrict__ Vt)
{
  const int c   = blockIdx.x * 256 + threadIdx.x;   // 0..D-1
  const int b   = blockIdx.y >> 7;                  // /(SC/8)
  const int kv0 = (blockIdx.y & 127) * 8;
  v8h pk;
#pragma unroll
  for (int j = 0; j < 8; ++j)
    pk[j] = (_Float16)src[((size_t)b * SEQ_C + kv0 + j) * D_MODEL + c];
  *(v8h*)(Vt + ((size_t)b * D_MODEL + c) * SKV + kv0) = pk;
}

// =====================================================================
// GEMM: C = oscale * (A[M,K] @ W[N,K]^T + bias[N]).  W f16.
// OUTMODE: 0 = fp32 row-major, 1 = f16 row-major,
//          2 = f16 into Vt[b][col][SC + q] (V projection, packed b128).
// Block = 8 waves = 32(M) x 512(N); A tile (32x32) async-staged to LDS,
// B fragment reused for both M sub-tiles.  grid = (N/512, M/32).
// =====================================================================
template <typename TA, int OUTMODE>
__global__ __launch_bounds__(256) void gemm_xwT_bias(
    const TA* __restrict__ A, const _Float16* __restrict__ W,
    const float* __restrict__ bias, void* __restrict__ Cout,
    int M, int N, int K, float oscale)
{
  constexpr int ASTR = (sizeof(TA) == 4) ? 36 : 40;  // bank-conflict-free pads
  const int lane = threadIdx.x & 31;
  const int wave = threadIdx.x >> 5;
  const int hl   = lane >> 4;
  const int ln   = lane & 15;
  const int mbase = blockIdx.y * 32;
  const int nbase = (blockIdx.x * 8 + wave) * 64;

  __shared__ __align__(16) TA ash[2][32 * ASTR];

  // async-copy assignment: 256 threads x 4 elements = 32x32 tile
  const int tid    = threadIdx.x;
  const int arow_c = (tid * 4) >> 5;       // 0..31
  const int acol_c = (tid * 4) & 31;       // multiple of 4
  const TA*      asrc  = A + (size_t)(mbase + arow_c) * K + acol_c;
  const uint32_t adst0 = lds_off(&ash[0][arow_c * ASTR + acol_c]);
  const uint32_t adst1 = lds_off(&ash[1][arow_c * ASTR + acol_c]);

  v8f acc[2][4] = {};

  if (sizeof(TA) == 4) async_b128(adst0, asrc); else async_b64(adst0, asrc);
  for (int k0 = 0; k0 < K; k0 += 32) {
    wait_async0();
    __syncthreads();
    const int buf = (k0 >> 5) & 1;
    if (k0 + 32 < K) {
      const TA* nsrc = asrc + k0 + 32;
      if (sizeof(TA) == 4) async_b128(buf ? adst0 : adst1, nsrc);
      else                 async_b64(buf ? adst0 : adst1, nsrc);
    }

    // ---- two A fragments from LDS (rows ln and 16+ln)
    v16h a[2];
#pragma unroll
    for (int m = 0; m < 2; ++m) {
      const TA* ar = &ash[buf][(m * 16 + ln) * ASTR + hl * 8];
      if (sizeof(TA) == 2) {
        a[m] = cat8(*(const v8h*)ar, *(const v8h*)(ar + 16));
      } else {
#pragma unroll
        for (int i = 0; i < 8; ++i) {
          a[m][i]     = (_Float16)ar[i];
          a[m][8 + i] = (_Float16)ar[16 + i];
        }
      }
    }

    // ---- B fragments: f16 weights load straight into the WMMA operand
#pragma unroll
    for (int t = 0; t < 4; ++t) {
      const _Float16* wrow =
          W + (size_t)(nbase + t * 16 + ln) * K + k0 + hl * 8;
      if (k0 + 32 < K) __builtin_prefetch(wrow + 32, 0, 1);
      const v16h bf = cat8(*(const v8h*)wrow, *(const v8h*)(wrow + 16));
#pragma unroll
      for (int m = 0; m < 2; ++m)
        acc[m][t] = __builtin_amdgcn_wmma_f32_16x16x32_f16(
            false, a[m], false, bf, (short)0, acc[m][t], false, false);
    }
  }

  // ---- epilogue
#pragma unroll
  for (int t = 0; t < 4; ++t) {
    const int col = nbase + t * 16 + ln;
    const float bcol = bias[col];
#pragma unroll
    for (int m = 0; m < 2; ++m) {
      if (OUTMODE == 2) {
        // V projection: 8 consecutive q per lane -> one packed b128 store
        const int row0 = mbase + m * 16 + hl * 8;
        const int bb = row0 >> 10;          // batch (SQ = 1024)
        const int q0 = row0 & 1023;
        v8h pk;
#pragma unroll
        for (int r = 0; r < 8; ++r)
          pk[r] = (_Float16)((acc[m][t][r] + bcol) * oscale);
        *(v8h*)((_Float16*)Cout +
                ((size_t)bb * D_MODEL + col) * SKV + SEQ_C + q0) = pk;
      } else {
#pragma unroll
        for (int r = 0; r < 8; ++r) {
          const int row = mbase + m * 16 + r + hl * 8;
          const float v = (acc[m][t][r] + bcol) * oscale;
          if (OUTMODE == 1) ((_Float16*)Cout)[(size_t)row * N + col] = (_Float16)v;
          else              ((float*)Cout)[(size_t)row * N + col] = v;
        }
      }
    }
  }
}

// =====================================================================
// Flash attention with KV cache, block-cooperative, all-f16 data path.
// Q pre-scaled by 1/sqrt(hd).  K tile staged row-major [kv][d]; V tile
// staged from transposed Vt as [d][kv] so BOTH K and V WMMA operands are
// 2 x ds_load_b128.  Causal masking only on boundary tiles.
// grid = (B*H, SQ/64), block = 128.
// =====================================================================
#define KSTR 136   // K tile row stride (halfs)
#define VTSTR 40   // V tile kv-stride (halfs)

__global__ __launch_bounds__(128) void attn_flash(
    const _Float16* __restrict__ Qp, const _Float16* __restrict__ Kp,
    const _Float16* __restrict__ cachedK, const _Float16* __restrict__ Vt,
    _Float16* __restrict__ O)
{
  const int lane = threadIdx.x & 31;
  const int wave = threadIdx.x >> 5;
  const int hl   = lane >> 4;
  const int ln   = lane & 15;
  const int b    = blockIdx.x / NUM_HEADS;
  const int h    = blockIdx.x % NUM_HEADS;
  const int qblk  = blockIdx.y * 64;
  const int qbase = qblk + wave * 16;

  __shared__ __align__(16) _Float16 ksh[2][32 * KSTR];
  __shared__ __align__(16) _Float16 vsh[2][128 * VTSTR];
  __shared__ __align__(16) _Float16 pbuf_all[4][16 * 32];
  _Float16* pbuf = pbuf_all[wave];

  // ---- Q fragments (once)
  v16h qf[4];
  const _Float16* qrow =
      Qp + ((size_t)b * SEQ_Q + qbase + ln) * D_MODEL + h * HEAD_DIM;
#pragma unroll
  for (int f = 0; f < 4; ++f) {
    const _Float16* p0 = qrow + f * 32 + hl * 8;
    qf[f] = cat8(*(const v8h*)p0, *(const v8h*)(p0 + 16));
  }

  v8f oacc[8] = {};
  float mrow[8], lsum[8];
#pragma unroll
  for (int r = 0; r < 8; ++r) { mrow[r] = -__builtin_inff(); lsum[r] = 0.f; }

  const int my_steps  = (SEQ_C + qbase + 16 + 31) / 32;
  const int blk_steps = (SEQ_C + qblk + 64) / 32;

  // async-copy assignment
  const int tid  = threadIdx.x;
  const int crow = tid >> 2;                 // K: kv row 0..31
  const int cseg = (tid & 3) * 32;           // K: d segment
  const _Float16* vrow_src =
      Vt + ((size_t)b * D_MODEL + h * HEAD_DIM + tid) * SKV;  // V: d row = tid

  auto issue_tile = [&](int s, int buf) {
    const int kv = s * 32 + crow;
    const _Float16* kb = (kv < SEQ_C)
        ? cachedK + ((size_t)b * SEQ_C + kv) * D_MODEL + h * HEAD_DIM
        : Kp + ((size_t)b * SEQ_Q + (kv - SEQ_C)) * D_MODEL + h * HEAD_DIM;
    const uint32_t kdst = lds_off(&ksh[buf][crow * KSTR + cseg]);
    const uint32_t vdst = lds_off(&vsh[buf][tid * VTSTR]);
    const _Float16* vsrc = vrow_src + s * 32;
#pragma unroll
    for (int j = 0; j < 4; ++j) {            // 32 halfs = 4 x b128 each
      async_b128(kdst + j * 16, kb + cseg + j * 8);
      async_b128(vdst + j * 16, vsrc + j * 8);
    }
  };

  issue_tile(0, 0);
  for (int s = 0; s < blk_steps; ++s) {
    wait_async0();
    __syncthreads();                                    // step-s tiles visible
    if (s + 1 < blk_steps) issue_tile(s + 1, (s + 1) & 1);

    if (s < my_steps) {
      const int kvbase = s * 32;
      const _Float16* kt = ksh[s & 1];
      const _Float16* vt = vsh[s & 1];

      // ------- S = Q K^T (two 16x16 accumulators over kv tile of 32)
      v8f s0 = {}, s1 = {};
#pragma unroll
      for (int f = 0; f < 4; ++f) {
        const _Float16* q0 = kt + ln * KSTR + f * 32 + hl * 8;
        const _Float16* q1 = kt + (16 + ln) * KSTR + f * 32 + hl * 8;
        const v16h kb0 = cat8(*(const v8h*)q0, *(const v8h*)(q0 + 16));
        const v16h kb1 = cat8(*(const v8h*)q1, *(const v8h*)(q1 + 16));
        s0 = __builtin_amdgcn_wmma_f32_16x16x32_f16(false, qf[f], false, kb0,
                                                    (short)0, s0, false, false);
        s1 = __builtin_amdgcn_wmma_f32_16x16x32_f16(false, qf[f], false, kb1,
                                                    (short)0, s1, false, false);
      }

      // ------- online softmax; rows live in half-waves (row = r + hl*8)
      float alpha[8];
      auto softmax_body = [&](bool domask) {
#pragma unroll
        for (int r = 0; r < 8; ++r) {
          const int qi = qbase + r + hl * 8;
          float v0 = s0[r];
          float v1 = s1[r];
          if (domask) {
            if (kvbase + ln      > SEQ_C + qi) v0 = -__builtin_inff();
            if (kvbase + 16 + ln > SEQ_C + qi) v1 = -__builtin_inff();
          }
          float rm = fmaxf(v0, v1);
#pragma unroll
          for (int off = 1; off < 16; off <<= 1)
            rm = fmaxf(rm, __shfl_xor(rm, off, 32));
          const float nm = fmaxf(mrow[r], rm);
          alpha[r] = __expf(mrow[r] - nm);
          float p0, p1;
          if (domask) {
            p0 = (v0 == -__builtin_inff()) ? 0.f : __expf(v0 - nm);
            p1 = (v1 == -__builtin_inff()) ? 0.f : __expf(v1 - nm);
          } else {
            p0 = __expf(v0 - nm);
            p1 = __expf(v1 - nm);
          }
          float rs = p0 + p1;
#pragma unroll
          for (int off = 1; off < 16; off <<= 1)
            rs += __shfl_xor(rs, off, 32);
          lsum[r] = lsum[r] * alpha[r] + rs;
          mrow[r] = nm;
          pbuf[(r + hl * 8) * 32 + ln]      = (_Float16)p0;
          pbuf[(r + hl * 8) * 32 + 16 + ln] = (_Float16)p1;
        }
      };
      const bool needmask = (kvbase + 31 > SEQ_C + qbase);  // wave-uniform
      if (needmask) softmax_body(true); else softmax_body(false);

#pragma unroll
      for (int t = 0; t < 8; ++t)
#pragma unroll
        for (int r = 0; r < 8; ++r) oacc[t][r] *= alpha[r];

      // wave-internal LDS store->load ordering
      __asm__ __volatile__("" ::: "memory");
      __builtin_amdgcn_wave_barrier();

      // ------- P as A-fragment (16x32 f16)
      const _Float16* pr = pbuf + ln * 32 + hl * 8;
      const v16h pa = cat8(*(const v8h*)pr, *(const v8h*)(pr + 16));

      // ------- O += P @ V: B operand now 2 x ds_load_b128 per sub-tile
#pragma unroll
      for (int t = 0; t < 8; ++t) {
        const _Float16* vp = vt + (t * 16 + ln) * VTSTR + hl * 8;
        const v16h vb = cat8(*(const v8h*)vp, *(const v8h*)(vp + 16));
        oacc[t] = __builtin_amdgcn_wmma_f32_16x16x32_f16(
            false, pa, false, vb, (short)0, oacc[t], false, false);
      }
    }
  }

  // ---- finalize: divide by l, store f16 as [B, SQ, H, hd]
  float inv[8];
#pragma unroll
  for (int r = 0; r < 8; ++r) inv[r] = 1.f / lsum[r];
#pragma unroll
  for (int t = 0; t < 8; ++t) {
    const int d = h * HEAD_DIM + t * 16 + ln;
#pragma unroll
    for (int r = 0; r < 8; ++r) {
      const int q = qbase + r + hl * 8;
      O[((size_t)b * SEQ_Q + q) * D_MODEL + d] = (_Float16)(oacc[t][r] * inv[r]);
    }
  }
}

// =====================================================================
extern "C" void kernel_launch(void* const* d_in, const int* in_sizes, int n_in,
                              void* d_out, int out_size, void* d_ws, size_t ws_size,
                              hipStream_t stream) {
  const float* query    = (const float*)d_in[0];
  const float* key      = (const float*)d_in[1];
  const float* value    = (const float*)d_in[2];
  const float* cached_k = (const float*)d_in[3];
  const float* cached_v = (const float*)d_in[4];
  const float* Wq = (const float*)d_in[5];
  const float* bq = (const float*)d_in[6];
  const float* Wk = (const float*)d_in[7];
  const float* bk = (const float*)d_in[8];
  const float* Wv = (const float*)d_in[9];
  const float* bv = (const float*)d_in[10];
  const float* Wo = (const float*)d_in[11];
  const float* bo = (const float*)d_in[12];
  float* out = (float*)d_out;

  const size_t WEL = (size_t)D_MODEL * D_MODEL;        // 4.19M
  const size_t AEL = (size_t)BATCH * SEQ_Q * D_MODEL;  // 8.39M
  const size_t VTL = (size_t)BATCH * D_MODEL * SKV;    // 16.78M
  _Float16* Wqh = (_Float16*)d_ws;
  _Float16* Wkh = Wqh + WEL;
  _Float16* Wvh = Wkh + WEL;
  _Float16* Woh = Wvh + WEL;
  _Float16* ckh = Woh + WEL;
  _Float16* Qph = ckh + AEL;
  _Float16* Kph = Qph + AEL;
  _Float16* Vt  = Kph + AEL;
  _Float16* Aoh = Vt  + VTL;                           // total 134.2 MB

  // ---- one-time conversions
  cvt_f32_to_f16<<<WEL / 2048, 256, 0, stream>>>(Wq, Wqh, (int)WEL);
  cvt_f32_to_f16<<<WEL / 2048, 256, 0, stream>>>(Wk, Wkh, (int)WEL);
  cvt_f32_to_f16<<<WEL / 2048, 256, 0, stream>>>(Wv, Wvh, (int)WEL);
  cvt_f32_to_f16<<<WEL / 2048, 256, 0, stream>>>(Wo, Woh, (int)WEL);
  cvt_f32_to_f16<<<AEL / 2048, 256, 0, stream>>>(cached_k, ckh, (int)AEL);
  cvt_transpose_v<<<dim3(D_MODEL / 256, BATCH * SEQ_C / 8), 256, 0, stream>>>(cached_v, Vt);

  const int M = BATCH * SEQ_Q;                         // 4096
  const float qscale = 0.08838834764831845f;           // 1/sqrt(HEAD_DIM)
  dim3 gblk(256);
  dim3 ggrd(D_MODEL / 512, M / 32);                    // (4, 128)

  // ---- QKV projections (Q pre-scaled; V written transposed into Vt)
  gemm_xwT_bias<float, 1><<<ggrd, gblk, 0, stream>>>(query, Wqh, bq, Qph, M, D_MODEL, D_MODEL, qscale);
  gemm_xwT_bias<float, 1><<<ggrd, gblk, 0, stream>>>(key,   Wkh, bk, Kph, M, D_MODEL, D_MODEL, 1.0f);
  gemm_xwT_bias<float, 2><<<ggrd, gblk, 0, stream>>>(value, Wvh, bv, Vt,  M, D_MODEL, D_MODEL, 1.0f);

  // ---- flash attention over cache + new KV
  dim3 ablk(128);
  dim3 agrd(BATCH * NUM_HEADS, SEQ_Q / 64);            // (64, 16)
  attn_flash<<<agrd, ablk, 0, stream>>>(Qph, Kph, ckh, Vt, Aoh);

  // ---- output projection (f16 in, fp32 out)
  gemm_xwT_bias<_Float16, 0><<<ggrd, gblk, 0, stream>>>(Aoh, Woh, bo, out, M, D_MODEL, D_MODEL, 1.0f);
}
